// PolicyNet_12833362280979
// MI455X (gfx1250) — matrix operand order repaired and checked by reading, not compiled, not run on hardware
//
#include <hip/hip_runtime.h>

// ---------------------------------------------------------------------------
// PolicyNet on gfx1250: conv(WMMA implicit GEMM) -> fc(WMMA) -> 50-step GRU
// (batch-tiled, register-resident h, bf16 WMMA with f32 accumulate,
//  async-to-LDS hidden-state staging).
// ---------------------------------------------------------------------------

typedef __attribute__((ext_vector_type(16))) __bf16 bf16x16;
typedef __attribute__((ext_vector_type(8)))  __bf16 bf16x8;
typedef __attribute__((ext_vector_type(8)))  float  f32x8;
typedef __attribute__((ext_vector_type(4)))  float  f32x4;

#define B_TOT   4096
#define CIN     60
#define CW      32
#define EMB     512
#define KFC     1152      // CW*6*6
#define KCV     540       // CIN*9
#define KCVP    576       // padded to 18*32
#define T_STEPS 50
#define BT      32        // batch tile per GRU block
#define XELEM   (B_TOT * CIN * 64)   // 15,728,640 elements of x

__device__ __forceinline__ float sigm_(float x) {
    return 1.0f / (1.0f + __expf(-x));
}
__device__ __forceinline__ float tanh_(float x) {
    x = fminf(fmaxf(x, -15.0f), 15.0f);
    float e = __expf(2.0f * x);
    return (e - 1.0f) / (e + 1.0f);
}

__device__ __forceinline__ f32x8 wmma_bf16(bf16x16 a, bf16x16 b, f32x8 c) {
    return __builtin_amdgcn_wmma_f32_16x16x32_bf16(
        false, a, false, b, (short)0, c, false, false);
}

__device__ __forceinline__ bf16x16 combine8(bf16x8 lo, bf16x8 hi) {
    bf16x16 a;
#pragma unroll
    for (int j = 0; j < 8; ++j) { a[j] = lo[j]; a[j + 8] = hi[j]; }
    return a;
}

// A-fragment 16x32 bf16 from row-major [M][K] buffer (global or LDS).
// lanes 0-15: M=lane, K chunks {kb..kb+7, kb+16..kb+23};
// lanes 16-31: M=lane-16, K chunks {kb+8..kb+15, kb+24..kb+31}.
__device__ __forceinline__ bf16x16 load_a16(const __bf16* base, int K, int kb, int lane) {
    const int kh = ((lane >> 4) & 1) << 3;   // 0 or 8
    const __bf16* p = base + (lane & 15) * K + kb + kh;
    bf16x8 lo = *(const bf16x8*)(p);
    bf16x8 hi = *(const bf16x8*)(p + 16);
    return combine8(lo, hi);
}

// B-fragment 32x16 bf16; wrow points at W + n_row*K (row-major [N][K] = B^T).
// lane n = lane&15, K chunks {kb+8h..kb+8h+7, kb+16+8h..kb+23+8h}, h = lane>>4.
__device__ __forceinline__ bf16x16 load_b16(const __bf16* wrow, int kb, int lane) {
    const __bf16* p = wrow + kb + (((lane >> 4) & 1) << 3);
    return combine8(*(const bf16x8*)(p), *(const bf16x8*)(p + 16));
}

// ---------------------------------------------------------------------------
// 0a) weight pre-conversion to bf16 + conv gather-offset table.
//     Pad table entries clamp to offset 0: the matching B (weight) entries are
//     zero, so WMMA multiplies the speculatively-loaded value by 0.0 -> no
//     branches needed in the conv gather.
// ---------------------------------------------------------------------------
__global__ void prep_kernel(const float* __restrict__ whh,
                            const float* __restrict__ fcw,
                            const float* __restrict__ cvw,
                            __bf16* __restrict__ whh_b,
                            __bf16* __restrict__ fcw_b,
                            __bf16* __restrict__ cvw_b,
                            int* __restrict__ tab) {
    int i = blockIdx.x * 256 + threadIdx.x;
    if (i < 3 * EMB * EMB)  whh_b[i] = (__bf16)whh[i];
    if (i < EMB * KFC)      fcw_b[i] = (__bf16)fcw[i];
    if (i < CW * KCVP) {
        int c = i / KCVP, k = i % KCVP;
        cvw_b[i] = (__bf16)(k < KCV ? cvw[c * KCV + k] : 0.0f);
    }
    if (i < KCVP) {
        if (i < KCV) {
            int cin = i / 9, r = i % 9;
            tab[i] = cin * 64 + (r / 3) * 8 + (r % 3);   // element offset in x[b]
        } else {
            tab[i] = 0;                                  // pad: weight is 0.0
        }
    }
}

// 0b) convert x to bf16 once (streaming, 8 elements / thread)
__global__ void xcvt_kernel(const float* __restrict__ x, __bf16* __restrict__ xb) {
    size_t i = ((size_t)blockIdx.x * 256 + threadIdx.x) * 8;
    if (i >= (size_t)XELEM) return;
    f32x4 a = *(const f32x4*)(x + i);
    f32x4 b = *(const f32x4*)(x + i + 4);
    bf16x8 o;
#pragma unroll
    for (int j = 0; j < 4; ++j) { o[j] = (__bf16)a[j]; o[j + 4] = (__bf16)b[j]; }
    *(bf16x8*)(xb + i) = o;
}

// ---------------------------------------------------------------------------
// 1) conv 3x3 VALID + relu as implicit GEMM per spatial position.
//    grid (B/16, 36), block 64 (2 waves, each one 16-channel N tile).
//    Unconditional bf16 gathers via offset table (pads killed by zero B).
//    act[b][c*36+s] bf16, matching the fc flatten order.
// ---------------------------------------------------------------------------
__global__ void conv_kernel(const __bf16* __restrict__ xbb,  // [B][CIN*64] bf16
                            const __bf16* __restrict__ wcv,  // [CW][576] bf16
                            const float* __restrict__ conv_b,
                            const int* __restrict__ tab,     // [576] gather offsets
                            __bf16* __restrict__ act) {
    const int lane  = threadIdx.x & 31;
    const int wv    = threadIdx.x >> 5;      // 0..1
    const int bbase = blockIdx.x * 16;
    const int s     = blockIdx.y;            // 0..35
    const int half = lane >> 4, nn = lane & 15;
    const int c = wv * 16 + nn;

    f32x8 acc;
    {
        float bias = conv_b[c];
#pragma unroll
        for (int i = 0; i < 8; ++i) acc[i] = bias;
    }
    const __bf16* wrow = wcv + c * KCVP;
    const int m = lane & 15;
    const int kh = half << 3;
    // fold the spatial origin (y0*8+x0 = s + 2*(s/6)) into the base pointer
    const __bf16* xb = xbb + (size_t)(bbase + m) * CIN * 64 + (s + 2 * (s / 6));

    for (int kt = 0; kt < 18; ++kt) {
        int kb = kt * 32;
        bf16x16 a;
#pragma unroll
        for (int j = 0; j < 16; ++j) {
            int k = kb + (j < 8 ? (kh + j) : (16 + kh + (j - 8)));
            a[j] = xb[tab[k]];               // unconditional gather
        }
        acc = wmma_bf16(a, load_b16(wrow, kb, lane), acc);
    }
#pragma unroll
    for (int i = 0; i < 8; ++i) {
        float v = acc[i] > 0.0f ? acc[i] : 0.0f;
        int b = bbase + i + half * 8;
        act[(size_t)b * KFC + c * 36 + s] = (__bf16)v;
    }
}

// ---------------------------------------------------------------------------
// 2) fc: emb = relu(act @ fc_w^T + fc_b), M=4096 N=512 K=1152.
//    grid (256,4), block 256 (8 waves, one 16-col N tile each).
//    Writes f32 emb (GRU register init) and bf16 emb (GRU async LDS init).
// ---------------------------------------------------------------------------
__global__ void fc_kernel(const __bf16* __restrict__ act,
                          const __bf16* __restrict__ fcw,   // [512][1152] bf16
                          const float* __restrict__ fc_b,
                          float* __restrict__ emb,
                          __bf16* __restrict__ embb) {
    const int lane  = threadIdx.x & 31;
    const int wv    = threadIdx.x >> 5;          // 0..7
    const int bbase = blockIdx.x * 16;
    const int half = lane >> 4, nn = lane & 15;
    const int e = (blockIdx.y * 8 + wv) * 16 + nn;

    f32x8 acc;
    {
        float bias = fc_b[e];
#pragma unroll
        for (int i = 0; i < 8; ++i) acc[i] = bias;
    }
    const __bf16* wrow = fcw + (size_t)e * KFC;
    const __bf16* arow = act + (size_t)bbase * KFC;

#pragma unroll 4
    for (int kt = 0; kt < KFC / 32; ++kt) {
        int kb = kt * 32;
        bf16x16 a = load_a16(arow, KFC, kb, lane);
        acc = wmma_bf16(a, load_b16(wrow, kb, lane), acc);
    }
#pragma unroll
    for (int i = 0; i < 8; ++i) {
        float v = acc[i] > 0.0f ? acc[i] : 0.0f;
        size_t idx = (size_t)(bbase + i + half * 8) * EMB + e;
        emb[idx]  = v;
        embb[idx] = (__bf16)v;
    }
}

// ---------------------------------------------------------------------------
// 3) GRU: 50 sequential steps, batch tile of 32 rows per block.
//    h kept fp32 in registers (D-fragment layout) + bf16 mirror in LDS.
//    16 waves; wave w owns e-tiles {2w, 2w+1} x both 16-row batch halves.
//    Per e-tile the 3 gate B-fragments are loaded ONCE and reused for both
//    batch halves (6 WMMA per 3 B loads) -> w_hh read exactly once per step.
//    Initial h staged into LDS with global_load_async_to_lds_b128.
// ---------------------------------------------------------------------------
__global__ __launch_bounds__(512)
void gru_kernel(const float* __restrict__ emb,     // f32  [B][512]
                const __bf16* __restrict__ embb,   // bf16 [B][512]
                const __bf16* __restrict__ whh,    // [1536][512] bf16
                const float* __restrict__ w_ih,    // [1536] (COR=1)
                const float* __restrict__ b_ih,    // [1536]
                const float* __restrict__ b_hh,    // [1536]
                const float* __restrict__ dec_w,   // [512]  (COR=1)
                const float* __restrict__ dec_b,   // [1]
                float* __restrict__ out) {         // [B][T]
    __shared__ __align__(16) __bf16 h_lds[BT * EMB];   // 32 KB
    __shared__ float outprev[BT];
    __shared__ float oacc[BT];

    const int lane  = threadIdx.x & 31;
    const int wv    = threadIdx.x >> 5;      // 0..15
    const int bbase = blockIdx.x * BT;
    const int half = lane >> 4, nn = lane & 15;

    // ---- async stage of initial hidden state (bf16) into LDS -------------
    {
        const __bf16* gsrc = embb + (size_t)bbase * EMB;   // 32 KB tile
        uint32_t lbase = (uint32_t)(uintptr_t)h_lds;
#pragma unroll
        for (int c = 0; c < 4; ++c) {
            int ch = c * 512 + threadIdx.x;                // 2048 x 16B chunks
            uint32_t loff = lbase + ch * 16;
            unsigned long long ga = (unsigned long long)(uintptr_t)(gsrc + ch * 8);
            asm volatile("global_load_async_to_lds_b128 %0, %1, off"
                         :: "v"(loff), "v"(ga) : "memory");
        }
        asm volatile("s_wait_asynccnt 0x0" ::: "memory");
    }

    // per-wave tiles: q = 2*p + mt ; p selects e-tile (et = 2*wv + p)
    f32x8 hreg[4];
    int   e_q[2];
    float cr_q[2], cz_q[2], cnb_q[2], gnb_q[2];
    float wir_q[2], wiz_q[2], win_q[2], dw_q[2];

#pragma unroll
    for (int p = 0; p < 2; ++p) {
        int et = 2 * wv + p;
        int e  = et * 16 + nn;
        e_q[p] = e;
        wir_q[p] = w_ih[e];
        wiz_q[p] = w_ih[EMB + e];
        win_q[p] = w_ih[2 * EMB + e];
        cr_q[p]  = b_ih[e] + b_hh[e];
        cz_q[p]  = b_ih[EMB + e] + b_hh[EMB + e];
        cnb_q[p] = b_hh[2 * EMB + e];          // inside r*gh_n
        gnb_q[p] = b_ih[2 * EMB + e];          // outside (gi_n)
        dw_q[p]  = dec_w[e];
        // initial hidden state fp32 registers (D-fragment layout)
#pragma unroll
        for (int mt = 0; mt < 2; ++mt)
#pragma unroll
            for (int i = 0; i < 8; ++i) {
                int rl = mt * 16 + i + half * 8;
                hreg[2 * p + mt][i] = emb[(size_t)(bbase + rl) * EMB + e];
            }
    }
    if (threadIdx.x < BT) { outprev[threadIdx.x] = 0.0f; oacc[threadIdx.x] = 0.0f; }
    __syncthreads();

    const float decb = dec_b[0];

    for (int ts = 0; ts < T_STEPS; ++ts) {
        float pf[2][8];
#pragma unroll
        for (int m = 0; m < 2; ++m)
#pragma unroll
            for (int i = 0; i < 8; ++i) pf[m][i] = 0.0f;

#pragma unroll
        for (int p = 0; p < 2; ++p) {
            const int e = e_q[p];
            const __bf16* wr_r = whh + (size_t)e * EMB;
            const __bf16* wr_z = whh + (size_t)(EMB + e) * EMB;
            const __bf16* wr_n = whh + (size_t)(2 * EMB + e) * EMB;

            f32x8 ar0, az0, an0, ar1, az1, an1;
#pragma unroll
            for (int i = 0; i < 8; ++i) {
                float op0 = outprev[i + half * 8];
                float op1 = outprev[16 + i + half * 8];
                ar0[i] = cr_q[p] + op0 * wir_q[p];
                az0[i] = cz_q[p] + op0 * wiz_q[p];
                an0[i] = cnb_q[p];
                ar1[i] = cr_q[p] + op1 * wir_q[p];
                az1[i] = cz_q[p] + op1 * wiz_q[p];
                an1[i] = cnb_q[p];
            }
#pragma unroll 4
            for (int kt = 0; kt < EMB / 32; ++kt) {
                int kb = kt * 32;
                bf16x16 br = load_b16(wr_r, kb, lane);
                bf16x16 bz = load_b16(wr_z, kb, lane);
                bf16x16 bn = load_b16(wr_n, kb, lane);
                bf16x16 a0 = load_a16(h_lds, EMB, kb, lane);
                bf16x16 a1 = load_a16(h_lds + 16 * EMB, EMB, kb, lane);
                ar0 = wmma_bf16(a0, br, ar0);
                az0 = wmma_bf16(a0, bz, az0);
                an0 = wmma_bf16(a0, bn, an0);
                ar1 = wmma_bf16(a1, br, ar1);
                az1 = wmma_bf16(a1, bz, az1);
                an1 = wmma_bf16(a1, bn, an1);
            }
            // gating for both batch halves of this e-tile
#pragma unroll
            for (int mt = 0; mt < 2; ++mt) {
                f32x8& ar = mt ? ar1 : ar0;
                f32x8& az = mt ? az1 : az0;
                f32x8& an = mt ? an1 : an0;
#pragma unroll
                for (int i = 0; i < 8; ++i) {
                    float op = outprev[mt * 16 + i + half * 8];
                    float r  = sigm_(ar[i]);
                    float z  = sigm_(az[i]);
                    float n  = tanh_(gnb_q[p] + op * win_q[p] + r * an[i]);
                    float hn = (1.0f - z) * n + z * hreg[2 * p + mt][i];
                    hreg[2 * p + mt][i] = hn;
                    pf[mt][i] += hn * dw_q[p];   // decoder partial (COR=1 dot)
                }
            }
        }
        __syncthreads();   // all WMMA reads of h_lds done

        // publish hnew (bf16) for next step's A-fragments
#pragma unroll
        for (int p = 0; p < 2; ++p) {
            const int e = e_q[p];
#pragma unroll
            for (int mt = 0; mt < 2; ++mt)
#pragma unroll
                for (int i = 0; i < 8; ++i)
                    h_lds[(mt * 16 + i + half * 8) * EMB + e] =
                        (__bf16)hreg[2 * p + mt][i];
        }
        // decoder reduction: lanes of each 16-group hold the same 8 rows ->
        // xor-shuffle reduce within group, then one LDS atomic.
#pragma unroll
        for (int m = 0; m < 2; ++m) {
#pragma unroll
            for (int i = 0; i < 8; ++i) {
                float v = pf[m][i];
                v += __shfl_xor(v, 1, 32);
                v += __shfl_xor(v, 2, 32);
                v += __shfl_xor(v, 4, 32);
                v += __shfl_xor(v, 8, 32);
                if (nn == 0) atomicAdd(&oacc[m * 16 + i + half * 8], v);
            }
        }
        __syncthreads();

        if (threadIdx.x < BT) {
            float o = oacc[threadIdx.x] + decb;
            out[(size_t)(bbase + threadIdx.x) * T_STEPS + ts] = o;
            outprev[threadIdx.x] = o;   // fed back as next-step input
            oacc[threadIdx.x]    = 0.0f;
        }
        __syncthreads();
    }
}

// ---------------------------------------------------------------------------
// launch
// ---------------------------------------------------------------------------
extern "C" void kernel_launch(void* const* d_in, const int* in_sizes, int n_in,
                              void* d_out, int out_size, void* d_ws, size_t ws_size,
                              hipStream_t stream) {
    (void)in_sizes; (void)n_in; (void)out_size; (void)ws_size;
    const float* x      = (const float*)d_in[0];
    const float* conv_w = (const float*)d_in[1];
    const float* conv_b = (const float*)d_in[2];
    const float* fc_w   = (const float*)d_in[3];
    const float* fc_b   = (const float*)d_in[4];
    const float* w_ih   = (const float*)d_in[5];
    const float* b_ih   = (const float*)d_in[6];
    const float* w_hh   = (const float*)d_in[7];
    const float* b_hh   = (const float*)d_in[8];
    const float* dec_w  = (const float*)d_in[9];
    const float* dec_b  = (const float*)d_in[10];

    char* ws = (char*)d_ws;
    size_t o = 0;
    __bf16* whh_b = (__bf16*)(ws + o); o += (size_t)3 * EMB * EMB * 2;   // 1.5 MB
    __bf16* fcw_b = (__bf16*)(ws + o); o += (size_t)EMB * KFC * 2;       // 1.1 MB
    __bf16* cvw_b = (__bf16*)(ws + o); o += (size_t)CW * KCVP * 2;       // 36 KB
    __bf16* act_b = (__bf16*)(ws + o); o += (size_t)B_TOT * KFC * 2;     // 9.4 MB
    float*  embf  = (float*)(ws + o);  o += (size_t)B_TOT * EMB * 4;     // 8.4 MB
    __bf16* embb  = (__bf16*)(ws + o); o += (size_t)B_TOT * EMB * 2;     // 4.2 MB
    __bf16* xbb   = (__bf16*)(ws + o); o += (size_t)XELEM * 2;           // 31.5 MB
    int*    tab   = (int*)(ws + o);                                      // 2.3 KB

    prep_kernel<<<(3 * EMB * EMB + 255) / 256, 256, 0, stream>>>(
        w_hh, fc_w, conv_w, whh_b, fcw_b, cvw_b, tab);
    xcvt_kernel<<<(XELEM / 8 + 255) / 256, 256, 0, stream>>>(x, xbb);
    conv_kernel<<<dim3(B_TOT / 16, 36), 64, 0, stream>>>(xbb, cvw_b, conv_b, tab, act_b);
    fc_kernel<<<dim3(B_TOT / 16, 4), 256, 0, stream>>>(act_b, fcw_b, fc_b, embf, embb);
    gru_kernel<<<B_TOT / BT, 512, 0, stream>>>(
        embf, embb, whh_b, w_ih, b_ih, b_hh, dec_w, dec_b, (float*)d_out);
}
